// ImageClassifier_68444598829695
// MI455X (gfx1250) — 8-bit matrix operand order repaired and checked by reading, compiled, not run on hardware
//
#include <hip/hip_runtime.h>
#include <stdint.h>


typedef __attribute__((ext_vector_type(8))) int   v8i;
typedef __attribute__((ext_vector_type(8))) float v8f;
typedef __attribute__((ext_vector_type(2))) float v2f;

__device__ __forceinline__ float bn_apply(const float* __restrict__ bn, int C, int c, float f) {
    float g  = bn[c];
    float be = bn[C + c];
    float mu = bn[2 * C + c];
    float va = bn[3 * C + c];
    return g * (f - mu) * rsqrtf(va + 1e-5f) + be;
}

__global__ void conv1_kernel(const float* __restrict__ x, const float* __restrict__ w1,
                             const float* __restrict__ b1, const float* __restrict__ bn1,
                             uint8_t* __restrict__ out, int B) {
    int p = blockIdx.x * blockDim.x + threadIdx.x;
    int M = B * 900;
    if (p >= M) return;
    int b = p / 900, r = p % 900, oh = r / 30, ow = r % 30;
    float acc[16];
#pragma unroll
    for (int co = 0; co < 16; ++co) acc[co] = 0.f;
    for (int ci = 0; ci < 3; ++ci)
#pragma unroll
        for (int kh = 0; kh < 3; ++kh)
#pragma unroll
            for (int kw = 0; kw < 3; ++kw) {
                float xv = x[((size_t)(b * 3 + ci) * 32 + (oh + kh)) * 32 + (ow + kw)];
                int   wo = (ci * 3 + kh) * 3 + kw;
#pragma unroll
                for (int co = 0; co < 16; ++co)
                    acc[co] = fmaf(xv, w1[co * 27 + wo], acc[co]);
            }
    size_t o = ((size_t)(b * 30 + oh) * 30 + ow) * 16;
#pragma unroll
    for (int co = 0; co < 16; ++co) {
        float y = bn_apply(bn1, 16, co, acc[co] + b1[co]);
        out[o + co] = (y > 0.f) ? (uint8_t)1 : (uint8_t)0;
    }
}

template <int Cin, int Cout>
__global__ void binprep_kernel(const float* __restrict__ w, uint32_t* __restrict__ wfrag) {
    constexpr int Ktot = 9 * Cin;
    constexpr int Kch  = (Ktot + 63) / 64;
    constexpr int Nt   = Cout / 16;
    int i = blockIdx.x * blockDim.x + threadIdx.x;
    if (i >= Nt * Kch * 256) return;
    int v = i & 7, lane = (i >> 3) & 31, ch = (i >> 8) % Kch, t = i / (Kch * 256);
    int hi = lane >> 4, n = t * 16 + (lane & 15);
    uint32_t packed = 0u;
#pragma unroll
    for (int bb = 0; bb < 4; ++bb) {
        int K = ch * 64 + (v >> 2) * 32 + hi * 16 + (v & 3) * 4 + bb;
        int8_t s = 0;
        if (K < Ktot) {
            int cell = K / Cin, cin = K % Cin, kh = cell / 3, kw = cell % 3;
            float wv = w[((n * Cin + cin) * 3 + kh) * 3 + kw];
            s = (wv > 0.f) ? (int8_t)1 : ((wv < 0.f) ? (int8_t)-1 : (int8_t)0);
        }
        packed |= ((uint32_t)(uint8_t)s) << (8 * bb);
    }
    wfrag[i] = packed;
}

template <int Cin, int Cout, int S, int Hin, int Win, int Hout, int Wout, bool OUTF>
__global__ void binconv_kernel(const uint8_t* __restrict__ in, const uint32_t* __restrict__ wfrag,
                               const float* __restrict__ bias, const float* __restrict__ bn,
                               void* __restrict__ outp, int B) {
    constexpr int Ktot = 9 * Cin;
    constexpr int Kch  = (Ktot + 63) / 64;
    constexpr int Nt   = Cout / 16;
    const int M    = B * Hout * Wout;
    const int lane = threadIdx.x & 31;
    const int tile = blockIdx.x * (blockDim.x >> 5) + (threadIdx.x >> 5);
    const int m0   = tile * 16;
    const int hi   = lane >> 4;

    int  p     = m0 + (lane & 15);
    bool valid = p < M;
    int  pc    = valid ? p : 0;
    int  b  = pc / (Hout * Wout), rr = pc % (Hout * Wout);
    int  oh = rr / Wout, ow = rr % Wout;
    int  ih0 = oh * S, iw0 = ow * S;

    v8i acc[Nt] = {};
    for (int ch = 0; ch < Kch; ++ch) {
        v8i a;
#pragma unroll
        for (int v = 0; v < 8; ++v) {
            int K = ch * 64 + (v >> 2) * 32 + hi * 16 + (v & 3) * 4;
            uint32_t val = 0u;
            if (valid && K < Ktot) {
                int cell = K / Cin, cin = K - cell * Cin;
                int kh = cell / 3, kw = cell - kh * 3;
                const uint8_t* pa =
                    in + ((size_t)(b * Hin + ih0 + kh) * Win + (iw0 + kw)) * Cin + cin;
                val = *(const uint32_t*)pa;
            }
            a[v] = (int)val;
        }
#pragma unroll
        for (int t = 0; t < Nt; ++t) {
            v8i bf = *(const v8i*)(wfrag + ((size_t)(t * Kch + ch) * 32 + lane) * 8);
            acc[t] = __builtin_amdgcn_wmma_i32_16x16x64_iu8(false, a, true, bf, acc[t],
                                                            false, false);
        }
    }

#pragma unroll
    for (int t = 0; t < Nt; ++t)
#pragma unroll
        for (int r = 0; r < 8; ++r) {
            int pm = m0 + r + 8 * hi;
            if (pm >= M) continue;
            int c  = t * 16 + (lane & 15);
            int b2 = pm / (Hout * Wout), r2 = pm % (Hout * Wout);
            int oh2 = r2 / Wout, ow2 = r2 % Wout;
            float f = (float)acc[t][r] + bias[c];
            float y = bn_apply(bn, Cout, c, f);
            size_t o = ((size_t)(b2 * Hout + oh2) * Wout + ow2) * Cout + c;
            if (OUTF) ((float*)outp)[o]   = fmaxf(y, 0.f);
            else      ((uint8_t*)outp)[o] = (y > 0.f) ? (uint8_t)1 : (uint8_t)0;
        }
}

__global__ void conv8prep_kernel(const float* __restrict__ w, float* __restrict__ wfrag) {
    int i = blockIdx.x * blockDim.x + threadIdx.x;
    if (i >= 4 * 144 * 32) return;
    int lane = i & 31, step = (i >> 5) % 144, t = i / (144 * 32);
    int hi = lane >> 4, n = t * 16 + (lane & 15);
#pragma unroll
    for (int q = 0; q < 2; ++q) {
        int K = step * 4 + hi * 2 + q;
        int cell = K / 64, cin = K % 64, kh = cell / 3, kw = cell % 3;
        wfrag[(size_t)i * 2 + q] = w[((n * 64 + cin) * 3 + kh) * 3 + kw];
    }
}

__global__ void conv8_kernel(const float* __restrict__ in, const float* __restrict__ wfrag,
                             const float* __restrict__ bias, const float* __restrict__ bn,
                             float* __restrict__ out, int B) {
    const int M    = B * 4;
    const int lane = threadIdx.x & 31;
    const int tile = blockIdx.x * (blockDim.x >> 5) + (threadIdx.x >> 5);
    const int m0   = tile * 16;
    const int hi   = lane >> 4;

    int  p     = m0 + (lane & 15);
    bool valid = p < M;
    int  pc    = valid ? p : 0;
    int  b = pc / 4, rr = pc % 4, oh = rr / 2, ow = rr % 2;

    v8f acc[4] = {};
    for (int step = 0; step < 144; ++step) {
        int K = step * 4 + hi * 2;
        int cell = K / 64, cin = K % 64, kh = cell / 3, kw = cell % 3;
        v2f av = {0.f, 0.f};
        if (valid) {
            const float* pa = in + ((size_t)(b * 4 + oh + kh) * 4 + (ow + kw)) * 64 + cin;
            av = *(const v2f*)pa;
        }
#pragma unroll
        for (int t = 0; t < 4; ++t) {
            v2f bv = *(const v2f*)(wfrag + ((size_t)(t * 144 + step) * 32 + lane) * 2);
            acc[t] = __builtin_amdgcn_wmma_f32_16x16x4_f32(false, av, false, bv,
                                                           (short)0, acc[t], false, false);
        }
    }
#pragma unroll
    for (int t = 0; t < 4; ++t)
#pragma unroll
        for (int r = 0; r < 8; ++r) {
            int pm = m0 + r + 8 * hi;
            if (pm >= M) continue;
            int c  = t * 16 + (lane & 15);
            int b2 = pm / 4, r2 = pm % 4, oh2 = r2 / 2, ow2 = r2 % 2;
            float y = bn_apply(bn, 64, c, acc[t][r] + bias[c]);
            out[((size_t)(b2 * 2 + oh2) * 2 + ow2) * 64 + c] = fmaxf(y, 0.f);
        }
}

__global__ void fc_kernel(const float* __restrict__ a8, const float* __restrict__ fw,
                          const float* __restrict__ fb, float* __restrict__ out, int B) {
    int idx = blockIdx.x * blockDim.x + threadIdx.x;
    if (idx >= B * 10) return;
    int b = idx / 10, j = idx % 10;
    const float* base = a8 + (size_t)b * 256;
    float s = fb[j];
    for (int c = 0; c < 64; ++c) {
        float feat = 0.25f * (base[c] + base[64 + c] + base[128 + c] + base[192 + c]);
        s = fmaf(feat, fw[j * 64 + c], s);
    }
    out[idx] = s;
}

extern "C" void kernel_launch(void* const* d_in, const int* in_sizes, int n_in,
                              void* d_out, int out_size, void* d_ws, size_t ws_size,
                              hipStream_t stream) {
    const float* x   = (const float*)d_in[0];
    const float* w1  = (const float*)d_in[1];  const float* b1 = (const float*)d_in[2];
    const float* w2  = (const float*)d_in[3];  const float* b2 = (const float*)d_in[4];
    const float* w3  = (const float*)d_in[5];  const float* b3 = (const float*)d_in[6];
    const float* w4  = (const float*)d_in[7];  const float* b4 = (const float*)d_in[8];
    const float* w5  = (const float*)d_in[9];  const float* b5 = (const float*)d_in[10];
    const float* w7  = (const float*)d_in[11]; const float* b7 = (const float*)d_in[12];
    const float* w8  = (const float*)d_in[13]; const float* b8 = (const float*)d_in[14];
    const float* bn1 = (const float*)d_in[15];
    const float* bn2 = (const float*)d_in[16];
    const float* bn3 = (const float*)d_in[17];
    const float* bn4 = (const float*)d_in[18];
    const float* bn5 = (const float*)d_in[19];
    const float* bn7 = (const float*)d_in[20];
    const float* bn8 = (const float*)d_in[21];
    const float* fcw = (const float*)d_in[22];
    const float* fcb = (const float*)d_in[23];
    const int B = in_sizes[0] / (3 * 32 * 32);

    char*  ws  = (char*)d_ws;
    size_t off = 0;
    auto carve = [&](size_t bytes) -> void* {
        void* pp = ws + off;
        off += (bytes + 255) & ~(size_t)255;
        return pp;
    };
    uint8_t*  bufX = (uint8_t*)carve((size_t)B * 30 * 30 * 16);
    uint8_t*  bufY = (uint8_t*)carve((size_t)B * 28 * 28 * 16);
    uint32_t* wf2  = (uint32_t*)carve(1 * 3 * 256 * 4);
    uint32_t* wf3  = (uint32_t*)carve(1 * 3 * 256 * 4);
    uint32_t* wf4  = (uint32_t*)carve(2 * 3 * 256 * 4);
    uint32_t* wf5  = (uint32_t*)carve(2 * 5 * 256 * 4);
    uint32_t* wf7  = (uint32_t*)carve(4 * 5 * 256 * 4);
    float*    wf8  = (float*)carve((size_t)4 * 144 * 32 * 2 * 4);

    binprep_kernel<16, 16><<<(1 * 3 * 256 + 255) / 256, 256, 0, stream>>>(w2, wf2);
    binprep_kernel<16, 16><<<(1 * 3 * 256 + 255) / 256, 256, 0, stream>>>(w3, wf3);
    binprep_kernel<16, 32><<<(2 * 3 * 256 + 255) / 256, 256, 0, stream>>>(w4, wf4);
    binprep_kernel<32, 32><<<(2 * 5 * 256 + 255) / 256, 256, 0, stream>>>(w5, wf5);
    binprep_kernel<32, 64><<<(4 * 5 * 256 + 255) / 256, 256, 0, stream>>>(w7, wf7);
    conv8prep_kernel<<<(4 * 144 * 32 + 255) / 256, 256, 0, stream>>>(w8, wf8);

    auto blocksFor = [](int M) { return ((M + 15) / 16 + 7) / 8; };

    const int M1 = B * 30 * 30;
    conv1_kernel<<<(M1 + 255) / 256, 256, 0, stream>>>(x, w1, b1, bn1, bufX, B);

    binconv_kernel<16, 16, 1, 30, 30, 28, 28, false>
        <<<blocksFor(B * 28 * 28), 256, 0, stream>>>(bufX, wf2, b2, bn2, bufY, B);
    binconv_kernel<16, 16, 1, 28, 28, 26, 26, false>
        <<<blocksFor(B * 26 * 26), 256, 0, stream>>>(bufY, wf3, b3, bn3, bufX, B);
    binconv_kernel<16, 32, 2, 26, 26, 12, 12, false>
        <<<blocksFor(B * 12 * 12), 256, 0, stream>>>(bufX, wf4, b4, bn4, bufY, B);
    binconv_kernel<32, 32, 1, 12, 12, 10, 10, false>
        <<<blocksFor(B * 10 * 10), 256, 0, stream>>>(bufY, wf5, b5, bn5, bufX, B);
    binconv_kernel<32, 64, 2, 10, 10, 4, 4, true>
        <<<blocksFor(B * 4 * 4), 256, 0, stream>>>(bufX, wf7, b7, bn7, bufY, B);

    conv8_kernel<<<blocksFor(B * 2 * 2), 256, 0, stream>>>((const float*)bufY, wf8, b8, bn8,
                                                           (float*)bufX, B);
    fc_kernel<<<(B * 10 + 255) / 256, 256, 0, stream>>>((const float*)bufX, fcw, fcb,
                                                        (float*)d_out, B);
}
